// CausalSelfAttention_16767552324137
// MI455X (gfx1250) — matrix operand order repaired and checked by reading, not compiled, not run on hardware
//
#include <hip/hip_runtime.h>
#include <hip/hip_bf16.h>

// ---------------------------------------------------------------------------
// CausalSelfAttention (B=2, T=2048, C=1024, H=16, Hkv=4, d=64) for gfx1250.
// All heavy math on v_wmma_f32_16x16x32_bf16 (fp32 accumulate), softmax/RoPE
// in fp32. Five kernels, all launched on `stream`, scratch carved from d_ws.
// ---------------------------------------------------------------------------

typedef __attribute__((ext_vector_type(16))) __bf16 v16bf;
typedef __attribute__((ext_vector_type(8)))  __bf16 v8bf;
typedef __attribute__((ext_vector_type(8)))  float  v8f;

#define B_    2
#define T_    2048
#define C_    1024
#define H_    16
#define HKV_  4
#define D_    64
#define NTOK  (B_ * T_)              // 4096 tokens
#define NQKV  (C_ + 2 * HKV_ * D_)   // 1536 fused qkv columns

__device__ __forceinline__ v8f wmma_bf16(v16bf a, v16bf b, v8f c) {
  // 8 args: (neg_a, A, neg_b, B, c_mod, C, reuse_a, reuse_b)
  return __builtin_amdgcn_wmma_f32_16x16x32_bf16(false, a, false, b, (short)0, c,
                                                 false, false);
}

// A-fragment: 16(M) x 32(K) bf16 from a row-major source.
// ISA layout: lanes 0-15 = rows; VGPR0-3 hold K = half*8 + 0..7,
// VGPR4-7 hold K = 16 + half*8 + 0..7 (half = lane>>4).
__device__ __forceinline__ v16bf load_frag_a(const __bf16* __restrict__ base, int ld,
                                             int row0, int k0, int lane) {
  const int r    = row0 + (lane & 15);
  const int half = (lane >> 4) & 1;
  const __bf16* p = base + (size_t)r * ld + k0 + half * 8;
  v8bf lo = *(const v8bf*)(p);        // K = half*8 + 0..7
  v8bf hi = *(const v8bf*)(p + 16);   // K = 16 + half*8 + 0..7
  v16bf f;
#pragma unroll
  for (int e = 0; e < 8; ++e) { f[e] = lo[e]; f[8 + e] = hi[e]; }
  return f;
}

// B-fragment: 32(K) x 16(N) bf16, loaded from the TRANSPOSED operand
// (N x K row-major). Elements 0-7 hold K = half*8 + 0..7, elements 8-15 hold
// K = 16 + half*8 + 0..7 (half = lane>>4), matching the A-fragment.
__device__ __forceinline__ v16bf load_frag_b(const __bf16* __restrict__ baseT, int ld,
                                             int n0, int k0, int lane) {
  const int n    = n0 + (lane & 15);
  const int half = (lane >> 4) & 1;
  const __bf16* p = baseT + (size_t)n * ld + k0 + half * 8;
  v8bf lo = *(const v8bf*)(p);
  v8bf hi = *(const v8bf*)(p + 16);
  v16bf f;
#pragma unroll
  for (int e = 0; e < 8; ++e) { f[e] = lo[e]; f[8 + e] = hi[e]; }
  return f;
}

__device__ __forceinline__ float halfmax16(float v) {
#pragma unroll
  for (int m = 1; m < 16; m <<= 1) v = fmaxf(v, __shfl_xor(v, m, 32));
  return v;
}
__device__ __forceinline__ float halfsum16(float v) {
#pragma unroll
  for (int m = 1; m < 16; m <<= 1) v += __shfl_xor(v, m, 32);
  return v;
}

// ---------------------------------------------------------------------------
// K0: cast x -> bf16; transpose+cast weights into Bt layouts (N x K).
// wqkvT rows: [0,1024)=wq^T, [1024,1280)=wk^T, [1280,1536)=wv^T.
// ---------------------------------------------------------------------------
__global__ void __launch_bounds__(256) convert_pack(
    const float* __restrict__ x,  const float* __restrict__ wq,
    const float* __restrict__ wk, const float* __restrict__ wv,
    const float* __restrict__ wp, __bf16* __restrict__ xb,
    __bf16* __restrict__ wqkvT,   __bf16* __restrict__ wpT) {
  const int NX = NTOK * C_;          // 4194304
  const int NQ = C_ * (H_ * D_);     // 1048576
  const int NK = C_ * (HKV_ * D_);   // 262144
  const int NV = C_ * (HKV_ * D_);   // 262144
  const int NP = C_ * C_;            // 1048576
  const int total = NX + NQ + NK + NV + NP;
  for (int i = blockIdx.x * blockDim.x + threadIdx.x; i < total;
       i += gridDim.x * blockDim.x) {
    if (i < NX) {
      xb[i] = (__bf16)x[i];
    } else if (i < NX + NQ) {
      int ii = i - NX; int kk = ii >> 10, n = ii & 1023;
      wqkvT[(size_t)n * C_ + kk] = (__bf16)wq[ii];
    } else if (i < NX + NQ + NK) {
      int ii = i - NX - NQ; int kk = ii >> 8, n = ii & 255;
      wqkvT[(size_t)(1024 + n) * C_ + kk] = (__bf16)wk[ii];
    } else if (i < NX + NQ + NK + NV) {
      int ii = i - NX - NQ - NK; int kk = ii >> 8, n = ii & 255;
      wqkvT[(size_t)(1280 + n) * C_ + kk] = (__bf16)wv[ii];
    } else {
      int ii = i - NX - NQ - NK - NV; int kk = ii >> 10, n = ii & 1023;
      wpT[(size_t)n * C_ + kk] = (__bf16)wp[ii];
    }
  }
}

// ---------------------------------------------------------------------------
// K1/K4: GEMM  C[M x N] (fp32) = A[M x K] (bf16) * Bt^T, Bt is N x K (bf16).
// Block = 8 waves (2 M x 4 N); each wave owns a 32x64 tile (8 accumulators),
// reusing each A fragment across 4 WMMAs. Block tile = 64 x 256.
// ---------------------------------------------------------------------------
__global__ void __launch_bounds__(256) gemm_bf16_f32(
    const __bf16* __restrict__ A, const __bf16* __restrict__ Bt,
    float* __restrict__ C, int M, int N, int K, int ldc) {
  const int lane = threadIdx.x & 31;
  const int w    = threadIdx.x >> 5;
  const int m0   = blockIdx.y * 64  + (w >> 2) * 32;
  const int n0   = blockIdx.x * 256 + (w & 3)  * 64;
  (void)M; (void)N;

  v8f acc[2][4] = {};
  for (int k0 = 0; k0 < K; k0 += 32) {
    if (k0 + 64 <= K)  // stream the next A chunk through L2 (global_prefetch_b8)
      __builtin_prefetch(A + (size_t)(m0 + (lane & 15)) * K + k0 + 64, 0, 1);
    v16bf a0 = load_frag_a(A, K, m0,      k0, lane);
    v16bf a1 = load_frag_a(A, K, m0 + 16, k0, lane);
#pragma unroll
    for (int n = 0; n < 4; ++n) {
      v16bf bfr = load_frag_b(Bt, K, n0 + n * 16, k0, lane);
      acc[0][n] = wmma_bf16(a0, bfr, acc[0][n]);
      acc[1][n] = wmma_bf16(a1, bfr, acc[1][n]);
    }
  }
  const int half = lane >> 4, ln = lane & 15;
#pragma unroll
  for (int i = 0; i < 2; ++i)
#pragma unroll
    for (int n = 0; n < 4; ++n)
#pragma unroll
      for (int r = 0; r < 8; ++r)
        C[(size_t)(m0 + i * 16 + r + 8 * half) * ldc + (n0 + n * 16 + ln)] =
            acc[i][n][r];
}

// ---------------------------------------------------------------------------
// K2: RoPE on q,k (fp32) + pack to bf16:
//   Qr[b][h][t][d], Kr[b][hkv][t][d], Vt[b][hkv][d][t] (V transposed so the
//   PV B-fragments are contiguous along keys).
// qkv layout: row = b*T + t, cols [0,1024)=q, [1024,1280)=k, [1280,1536)=v.
// ---------------------------------------------------------------------------
__global__ void __launch_bounds__(256) rope_pack(
    const float* __restrict__ qkv, __bf16* __restrict__ Qr,
    __bf16* __restrict__ Kr, __bf16* __restrict__ Vt) {
  const float LOG1E4_OVER_32 = 0.2878231366f;  // ln(10000)/32
  const int total = NTOK * NQKV;
  for (int i = blockIdx.x * blockDim.x + threadIdx.x; i < total;
       i += gridDim.x * blockDim.x) {
    const int n = i / NQKV, c = i - n * NQKV;  // n = b*T + t
    const int bb = n >> 11, t = n & (T_ - 1);
    const float val = qkv[i];
    if (c < 1024) {                       // ---- q + rope
      const int h = c >> 6, j = c & 63, jj = j & 31;
      const float theta = (float)t * __expf(-(float)jj * LOG1E4_OVER_32);
      float sn, cs; __sincosf(theta, &sn, &cs);
      const float other = qkv[(size_t)n * NQKV + (c ^ 32)];
      const float out = (j < 32) ? (val * cs - other * sn)
                                 : (val * cs + other * sn);
      Qr[((size_t)(bb * H_ + h) * T_ + t) * D_ + j] = (__bf16)out;
    } else if (c < 1280) {                // ---- k + rope
      const int cc = c - 1024;
      const int h = cc >> 6, j = cc & 63, jj = j & 31;
      const float theta = (float)t * __expf(-(float)jj * LOG1E4_OVER_32);
      float sn, cs; __sincosf(theta, &sn, &cs);
      const float other = qkv[(size_t)n * NQKV + (c ^ 32)];
      const float out = (j < 32) ? (val * cs - other * sn)
                                 : (val * cs + other * sn);
      Kr[((size_t)(bb * HKV_ + h) * T_ + t) * D_ + j] = (__bf16)out;
    } else {                              // ---- v, transposed
      const int cc = c - 1280;
      const int h = cc >> 6, j = cc & 63;
      Vt[((size_t)(bb * HKV_ + h) * D_ + j) * T_ + t] = (__bf16)val;
    }
  }
}

// ---------------------------------------------------------------------------
// K3: flash attention. One wave = one 16-query tile of one q head.
// Key tiles of 32: scores via 4 WMMAs, online softmax (half-wave shfl
// reductions matching the C/D layout: row M = r + 8*(lane>=16), col = lane&15),
// P re-shaped C-layout -> A-fragment through a per-wave LDS tile, PV via
// 4 WMMAs against V^T. Output written bf16 in (B,T,C) for the projection.
// ---------------------------------------------------------------------------
__global__ void __launch_bounds__(256) attn_fwd(
    const __bf16* __restrict__ Qr, const __bf16* __restrict__ Kr,
    const __bf16* __restrict__ Vt, __bf16* __restrict__ Y) {
  __shared__ __bf16 Plds[8][16 * 32];  // 1 KB per wave
  const int lane = threadIdx.x & 31;
  const int w    = threadIdx.x >> 5;
  const int idx  = blockIdx.x * 8 + w;       // [0, 4096)
  const int qt   = idx & 127;
  const int h    = (idx >> 7) & 15;
  const int b    = idx >> 11;
  const int hkv  = h >> 2;
  const int q0   = qt * 16;
  const int half = lane >> 4, ln = lane & 15;

  const __bf16* Qb = Qr + (size_t)(b * H_ + h)     * T_ * D_;
  const __bf16* Kb = Kr + (size_t)(b * HKV_ + hkv) * T_ * D_;
  const __bf16* Vb = Vt + (size_t)(b * HKV_ + hkv) * D_ * T_;

  const v16bf qa0 = load_frag_a(Qb, D_, q0, 0,  lane);   // d = 0..31
  const v16bf qa1 = load_frag_a(Qb, D_, q0, 32, lane);   // d = 32..63

  v8f acc[4] = {};
  float m_i[8], l_i[8];
#pragma unroll
  for (int r = 0; r < 8; ++r) { m_i[r] = -1e30f; l_i[r] = 0.0f; }

  const int ntiles = (q0 + 47) >> 5;  // key tiles covering s <= q0+15
  for (int jt = 0; jt < ntiles; ++jt) {
    const int kb = jt * 32;
    v8f s0 = {}, s1 = {};
    {  // scores: Q[16 x 64] * K^T -> two 16x16 tiles (keys kb.. / kb+16..)
      v16bf kf0 = load_frag_b(Kb, D_, kb,      0, lane);
      v16bf kf1 = load_frag_b(Kb, D_, kb + 16, 0, lane);
      s0 = wmma_bf16(qa0, kf0, s0);
      s1 = wmma_bf16(qa0, kf1, s1);
      kf0 = load_frag_b(Kb, D_, kb,      32, lane);
      kf1 = load_frag_b(Kb, D_, kb + 16, 32, lane);
      s0 = wmma_bf16(qa1, kf0, s0);
      s1 = wmma_bf16(qa1, kf1, s1);
    }
    const bool need_mask = (kb + 31) > q0;
#pragma unroll
    for (int r = 0; r < 8; ++r) {
      const int t = q0 + r + 8 * half;           // global query row
      float v0 = s0[r] * 0.125f;                 // 1/sqrt(64)
      float v1 = s1[r] * 0.125f;
      if (need_mask) {
        if (kb + ln      > t) v0 = -1e30f;       // causal: s > t
        if (kb + 16 + ln > t) v1 = -1e30f;
      }
      const float mr    = halfmax16(fmaxf(v0, v1));
      const float mnew  = fmaxf(m_i[r], mr);
      const float alpha = __expf(m_i[r] - mnew);
      const float p0 = __expf(v0 - mnew);
      const float p1 = __expf(v1 - mnew);
      l_i[r] = l_i[r] * alpha + halfsum16(p0 + p1);
      m_i[r] = mnew;
#pragma unroll
      for (int n = 0; n < 4; ++n) acc[n][r] *= alpha;
      const int row = r + 8 * half;              // stage P row-major in LDS
      Plds[w][row * 32 + ln]      = (__bf16)p0;
      Plds[w][row * 32 + 16 + ln] = (__bf16)p1;
    }
    asm volatile("s_wait_dscnt 0x0" ::: "memory");  // wave-local LDS RAW
    v16bf pa;  // reload P as an A-fragment (16 queries x 32 keys)
    {
      const __bf16* pp = &Plds[w][ln * 32 + half * 8];
      v8bf plo = *(const v8bf*)(pp);
      v8bf phi = *(const v8bf*)(pp + 16);
#pragma unroll
      for (int e = 0; e < 8; ++e) { pa[e] = plo[e]; pa[8 + e] = phi[e]; }
    }
#pragma unroll
    for (int n = 0; n < 4; ++n) {  // y += P * V  (V^T rows = output dims)
      v16bf vf = load_frag_b(Vb, T_, n * 16, kb, lane);
      acc[n] = wmma_bf16(pa, vf, acc[n]);
    }
  }
#pragma unroll
  for (int r = 0; r < 8; ++r) {
    const float inv = 1.0f / l_i[r];
    const int t = q0 + r + 8 * half;
    __bf16* yp = Y + (size_t)(b * T_ + t) * C_ + h * D_;
#pragma unroll
    for (int n = 0; n < 4; ++n) yp[n * 16 + ln] = (__bf16)(acc[n][r] * inv);
  }
}

// ---------------------------------------------------------------------------
// Host-side orchestration.
// ---------------------------------------------------------------------------
extern "C" void kernel_launch(void* const* d_in, const int* in_sizes, int n_in,
                              void* d_out, int out_size, void* d_ws, size_t ws_size,
                              hipStream_t stream) {
  (void)in_sizes; (void)n_in; (void)out_size; (void)ws_size;
  const float* x     = (const float*)d_in[0];
  const float* wq    = (const float*)d_in[1];
  const float* wk    = (const float*)d_in[2];
  const float* wv    = (const float*)d_in[3];
  const float* wproj = (const float*)d_in[4];

  char* ws = (char*)d_ws;
  const size_t SZ_XB   = (size_t)NTOK * C_ * 2;          //  8 MB  x bf16
  const size_t SZ_WQKV = (size_t)NQKV * C_ * 2;          //  3 MB  fused w^T bf16
  const size_t SZ_WP   = (size_t)C_ * C_ * 2;            //  2 MB  wproj^T bf16
  const size_t SZ_QKV  = (size_t)NTOK * NQKV * 4;        // 24 MB  qkv fp32
  const size_t SZ_QR   = (size_t)B_ * H_ * T_ * D_ * 2;  //  8 MB
  const size_t SZ_KR   = (size_t)B_ * HKV_ * T_ * D_ * 2;//  2 MB
  const size_t SZ_VT   = (size_t)B_ * HKV_ * D_ * T_ * 2;//  2 MB
  size_t off = 0;
  __bf16* xb    = (__bf16*)(ws + off); off += SZ_XB;
  __bf16* wqkvT = (__bf16*)(ws + off); off += SZ_WQKV;
  __bf16* wpT   = (__bf16*)(ws + off); off += SZ_WP;
  float*  qkv   = (float*) (ws + off); off += SZ_QKV;
  __bf16* Qrb   = (__bf16*)(ws + off); off += SZ_QR;
  __bf16* Krb   = (__bf16*)(ws + off); off += SZ_KR;
  __bf16* Vtb   = (__bf16*)(ws + off); off += SZ_VT;
  __bf16* Yb    = (__bf16*)(ws + off);                   // bf16 attn output

  // K0: casts + weight transposes
  convert_pack<<<2048, 256, 0, stream>>>(x, wq, wk, wv, wproj, xb, wqkvT, wpT);
  // K1: fused QKV projection (4096 x 1536 = 4096 x 1024 @ w^T)
  gemm_bf16_f32<<<dim3(NQKV / 256, NTOK / 64), 256, 0, stream>>>(
      xb, wqkvT, qkv, NTOK, NQKV, C_, NQKV);
  // K2: RoPE + pack Q/K/V^T to bf16
  rope_pack<<<2048, 256, 0, stream>>>(qkv, Qrb, Krb, Vtb);
  // K3: causal flash attention (512 blocks x 8 waves = 4096 query tiles)
  attn_fwd<<<(B_ * H_ * (T_ / 16)) / 8, 256, 0, stream>>>(Qrb, Krb, Vtb, Yb);
  // K4: output projection -> fp32 d_out
  gemm_bf16_f32<<<dim3(C_ / 256, NTOK / 64), 256, 0, stream>>>(
      Yb, wpT, (float*)d_out, NTOK, C_, C_, C_);
}
